// GCN_19576460935924
// MI455X (gfx1250) — matrix operand order repaired and checked by reading, not compiled, not run on hardware
//
#include <hip/hip_runtime.h>

typedef __bf16 bf16_t;
typedef __attribute__((ext_vector_type(16))) __bf16 v16bf;
typedef __attribute__((ext_vector_type(8)))  __bf16 v8bf;
typedef __attribute__((ext_vector_type(4)))  __bf16 v4bf;
typedef __attribute__((ext_vector_type(8)))  float  v8f;

#define HID 128
#define FIN 96
#define GR  512

// ---------------- utility kernels ----------------

__global__ void zero_f32(float* __restrict__ p, int n) {
  int i = blockIdx.x * blockDim.x + threadIdx.x;
  int s = gridDim.x * blockDim.x;
  for (; i < n; i += s) p[i] = 0.0f;
}

__global__ void deg_count(const long long* __restrict__ dst, float* __restrict__ deg, int E) {
  int e = blockIdx.x * blockDim.x + threadIdx.x;
  if (e < E) atomicAdd(&deg[(int)dst[e]], 1.0f);
}

// in-place: deg -> D^{-1/2} with self loop (deg+1)
__global__ void finalize_dinv(float* __restrict__ deg, int n) {
  int i = blockIdx.x * blockDim.x + threadIdx.x;
  if (i < n) deg[i] = rsqrtf(deg[i] + 1.0f);
}

__global__ void cvt_bf16(const float* __restrict__ in, bf16_t* __restrict__ out, int n) {
  int i = blockIdx.x * blockDim.x + threadIdx.x;
  int s = gridDim.x * blockDim.x;
  for (; i < n; i += s) out[i] = (bf16_t)in[i];
}

// W: [K x Ncols] row-major -> WT: [Ncols x K] bf16 (so B fragments are contiguous K-runs)
__global__ void transpose_bf16(const float* __restrict__ W, bf16_t* __restrict__ WT,
                               int K, int Ncols) {
  int i = blockIdx.x * blockDim.x + threadIdx.x;
  if (i < K * Ncols) {
    int k = i / Ncols, n = i % Ncols;
    WT[n * K + k] = (bf16_t)W[i];
  }
}

// ---------------- WMMA GEMM: C[M x 128] = A[M x K] * B[K x 128] ----------------
// A row-major bf16, BT = B transposed [128 x K] bf16, C f32.
// Block = 256 threads = 8 waves = 2 M-stripes x 4 waves; each wave computes TWO
// 16x16 N-tiles from ONE A fragment (2 WMMAs per k-step -> A reuse, fewer loads).
template <int K>
__global__ __launch_bounds__(256) void gemm_wmma(const bf16_t* __restrict__ A,
                                                 const bf16_t* __restrict__ BT,
                                                 float* __restrict__ C, int M) {
  const int lane    = threadIdx.x & 31;
  const int wave    = threadIdx.x >> 5;   // 0..7
  const int mstripe = wave >> 2;          // 0..1
  const int npair   = wave & 3;           // 0..3 -> N-tiles 2*npair, 2*npair+1
  const int m0      = (blockIdx.x * 2 + mstripe) * 16;
  const int n0      = npair * 32;
  const int row     = lane & 15;          // M-row (A) / N-col (B) within tile
  const int khalf   = lane >> 4;          // which K-half this lane holds
  const bool valid  = (m0 < M);           // M=50000 is a multiple of 16; tile-level guard

  int arow = m0 + row;
  if (arow >= M) arow = M - 1;            // clamp loads for the (rare) dead stripe

  // A frag (ISA 16-bit A 16x32): lane holds K = khalf*8 + [0..7] and 16 + khalf*8 + [0..7]
  const bf16_t* ap  = A  + (size_t)arow * K + khalf * 8;
  // B frags (16-bit B 32x16): lane holds col n + row, K = khalf*8 + [0..7] and 16 + khalf*8 + [0..7]
  const bf16_t* bp0 = BT + (size_t)(n0 + row) * K + khalf * 8;
  const bf16_t* bp1 = bp0 + (size_t)16 * K;

  v8f acc0 = {};
  v8f acc1 = {};
#pragma unroll
  for (int k0 = 0; k0 < K; k0 += 32) {
    v8bf alo = *(const v8bf*)(ap + k0);
    v8bf ahi = *(const v8bf*)(ap + k0 + 16);
    v16bf a = __builtin_shufflevector(alo, ahi, 0, 1, 2, 3, 4, 5, 6, 7,
                                      8, 9, 10, 11, 12, 13, 14, 15);
    v8bf b0lo = *(const v8bf*)(bp0 + k0);
    v8bf b0hi = *(const v8bf*)(bp0 + k0 + 16);
    v16bf b0 = __builtin_shufflevector(b0lo, b0hi, 0, 1, 2, 3, 4, 5, 6, 7,
                                       8, 9, 10, 11, 12, 13, 14, 15);
    v8bf b1lo = *(const v8bf*)(bp1 + k0);
    v8bf b1hi = *(const v8bf*)(bp1 + k0 + 16);
    v16bf b1 = __builtin_shufflevector(b1lo, b1hi, 0, 1, 2, 3, 4, 5, 6, 7,
                                       8, 9, 10, 11, 12, 13, 14, 15);
    acc0 = __builtin_amdgcn_wmma_f32_16x16x32_bf16(false, a, false, b0,
                                                   (short)0, acc0, false, false);
    acc1 = __builtin_amdgcn_wmma_f32_16x16x32_bf16(false, a, false, b1,
                                                   (short)0, acc1, false, false);
  }

  if (valid) {
    // D layout: VGPR v -> row m0 + v + (lane>=16)*8, col = tile_n0 + (lane&15)
    float* cp0 = C + (size_t)(m0 + khalf * 8) * HID + (n0 + row);
    float* cp1 = cp0 + 16;
#pragma unroll
    for (int v = 0; v < 8; ++v) cp0[v * HID] = acc0[v];
#pragma unroll
    for (int v = 0; v < 8; ++v) cp1[v * HID] = acc1[v];
  }
}

// ---------------- edge scatter-add: agg[dst] += h[src] * dinv[src]*dinv[dst] ----------------
// one wave per edge; each lane handles 4 contiguous features (float4, fully coalesced).
__global__ __launch_bounds__(256) void edge_agg(const float* __restrict__ h,
                                                const long long* __restrict__ src,
                                                const long long* __restrict__ dst,
                                                const float* __restrict__ dinv,
                                                float* __restrict__ agg, int E) {
  const int lane = threadIdx.x & 31;
  int w = (blockIdx.x * blockDim.x + threadIdx.x) >> 5;
  const int nw = (gridDim.x * blockDim.x) >> 5;
  for (int e = w; e < E; e += nw) {
    const int s = (int)src[e];
    const int d = (int)dst[e];
    const float norm = dinv[s] * dinv[d];
    const float4 v = ((const float4*)(h + (size_t)s * HID))[lane];
    float* ap = agg + (size_t)d * HID + lane * 4;
    atomicAdd(ap + 0, v.x * norm);
    atomicAdd(ap + 1, v.y * norm);
    atomicAdd(ap + 2, v.z * norm);
    atomicAdd(ap + 3, v.w * norm);
  }
}

// ---------------- layer-1 epilogue: h = relu(agg + lin/deg + b) -> bf16 ----------------
__global__ __launch_bounds__(256) void post_relu_bf16(const float* __restrict__ agg,
                                                      const float* __restrict__ lin,
                                                      const float* __restrict__ dinv,
                                                      const float* __restrict__ bias,
                                                      bf16_t* __restrict__ out, int N) {
  int t = blockIdx.x * blockDim.x + threadIdx.x;
  int node = t >> 5;
  if (node >= N) return;
  int f = (t & 31) * 4;
  float di = dinv[node];
  float sw = di * di;  // 1/deg
  float4 a = *(const float4*)(agg + (size_t)node * HID + f);
  float4 l = *(const float4*)(lin + (size_t)node * HID + f);
  float4 b = *(const float4*)(bias + f);
  v4bf o;
  o[0] = (bf16_t)fmaxf(a.x + l.x * sw + b.x, 0.0f);
  o[1] = (bf16_t)fmaxf(a.y + l.y * sw + b.y, 0.0f);
  o[2] = (bf16_t)fmaxf(a.z + l.z * sw + b.z, 0.0f);
  o[3] = (bf16_t)fmaxf(a.w + l.w * sw + b.w, 0.0f);
  *(v4bf*)(out + (size_t)node * HID + f) = o;
}

// ---------------- layer-2 epilogue + global mean pool (sum & count) ----------------
__global__ __launch_bounds__(256) void post_pool(const float* __restrict__ agg,
                                                 const float* __restrict__ lin,
                                                 const float* __restrict__ dinv,
                                                 const float* __restrict__ bias,
                                                 const long long* __restrict__ batch,
                                                 float* __restrict__ pool,
                                                 float* __restrict__ cnt, int N) {
  int t = blockIdx.x * blockDim.x + threadIdx.x;
  int node = t >> 5;
  if (node >= N) return;
  int f = (t & 31) * 4;
  float di = dinv[node];
  float sw = di * di;
  float4 a = *(const float4*)(agg + (size_t)node * HID + f);
  float4 l = *(const float4*)(lin + (size_t)node * HID + f);
  float4 b = *(const float4*)(bias + f);
  float v0 = fmaxf(a.x + l.x * sw + b.x, 0.0f);
  float v1 = fmaxf(a.y + l.y * sw + b.y, 0.0f);
  float v2 = fmaxf(a.z + l.z * sw + b.z, 0.0f);
  float v3 = fmaxf(a.w + l.w * sw + b.w, 0.0f);
  int g = (int)batch[node];
  float* p = pool + (size_t)g * HID + f;
  atomicAdd(p + 0, v0);
  atomicAdd(p + 1, v1);
  atomicAdd(p + 2, v2);
  atomicAdd(p + 3, v3);
  if ((t & 31) == 0) atomicAdd(&cnt[g], 1.0f);
}

// ---------------- head: out[g,:] = mean(pool[g]) @ Wlin + blin (one wave per graph) ----------------
__global__ __launch_bounds__(256) void head(const float* __restrict__ pool,
                                            const float* __restrict__ cnt,
                                            const float* __restrict__ Wlin,
                                            const float* __restrict__ blin,
                                            float* __restrict__ out) {
  int w = (blockIdx.x * blockDim.x + threadIdx.x) >> 5;
  int lane = threadIdx.x & 31;
  if (w >= GR) return;
  float invc = 1.0f / fmaxf(cnt[w], 1.0f);
  float d0 = 0.0f, d1 = 0.0f;
#pragma unroll
  for (int j = 0; j < 4; ++j) {
    int f = lane * 4 + j;
    float m = pool[(size_t)w * HID + f] * invc;
    d0 += m * Wlin[f * 2 + 0];
    d1 += m * Wlin[f * 2 + 1];
  }
#pragma unroll
  for (int off = 16; off > 0; off >>= 1) {
    d0 += __shfl_xor(d0, off, 32);
    d1 += __shfl_xor(d1, off, 32);
  }
  if (lane == 0) {
    out[w * 2 + 0] = d0 + blin[0];
    out[w * 2 + 1] = d1 + blin[1];
  }
}

// ---------------- launch ----------------

extern "C" void kernel_launch(void* const* d_in, const int* in_sizes, int n_in,
                              void* d_out, int out_size, void* d_ws, size_t ws_size,
                              hipStream_t stream) {
  (void)n_in; (void)out_size; (void)ws_size;
  const float*     x     = (const float*)d_in[0];
  const long long* ei    = (const long long*)d_in[1];
  const long long* batch = (const long long*)d_in[2];
  // d_in[3] = num_graphs scalar (fixed GR=512)
  const float* W1   = (const float*)d_in[4];
  const float* b1   = (const float*)d_in[5];
  const float* W2   = (const float*)d_in[6];
  const float* b2   = (const float*)d_in[7];
  const float* Wlin = (const float*)d_in[8];
  const float* blin = (const float*)d_in[9];
  float* out = (float*)d_out;

  const int N = in_sizes[0] / FIN;  // 50000
  const int E = in_sizes[1] / 2;    // 800000
  const long long* src = ei;
  const long long* dst = ei + E;

  // workspace carve-up (f32 region then bf16 region)
  float* wsf = (float*)d_ws;
  size_t off = 0;
  auto allocf = [&](size_t n) { float* p = wsf + off; off += (n + 63) & ~(size_t)63; return p; };
  float* dinv   = allocf((size_t)N);
  float* bufLin = allocf((size_t)N * HID);
  float* bufAgg = allocf((size_t)N * HID);
  float* pool   = allocf((size_t)GR * HID);
  float* cnt    = allocf((size_t)GR);
  bf16_t* wsb = (bf16_t*)(wsf + off);
  size_t offb = 0;
  auto allocb = [&](size_t n) { bf16_t* p = wsb + offb; offb += (n + 127) & ~(size_t)127; return p; };
  bf16_t* xb  = allocb((size_t)N * FIN);
  bf16_t* hb  = allocb((size_t)N * HID);
  bf16_t* W1T = allocb((size_t)FIN * HID);
  bf16_t* W2T = allocb((size_t)HID * HID);

  const int TB = 256;
  const int gblocks = (N + 31) / 32;  // 2 M-stripes per block

  // init + degree/normalization
  zero_f32<<<1024, TB, 0, stream>>>(dinv, N);
  zero_f32<<<256, TB, 0, stream>>>(pool, GR * HID);
  zero_f32<<<4, TB, 0, stream>>>(cnt, GR);
  deg_count<<<(E + TB - 1) / TB, TB, 0, stream>>>(dst, dinv, E);
  finalize_dinv<<<(N + TB - 1) / TB, TB, 0, stream>>>(dinv, N);

  // precision conversion / weight transpose
  cvt_bf16<<<2048, TB, 0, stream>>>(x, xb, N * FIN);
  transpose_bf16<<<(FIN * HID + TB - 1) / TB, TB, 0, stream>>>(W1, W1T, FIN, HID);
  transpose_bf16<<<(HID * HID + TB - 1) / TB, TB, 0, stream>>>(W2, W2T, HID, HID);

  // ---- layer 1 ----
  gemm_wmma<FIN><<<gblocks, TB, 0, stream>>>(xb, W1T, bufLin, N);
  zero_f32<<<2048, TB, 0, stream>>>(bufAgg, N * HID);
  edge_agg<<<2048, TB, 0, stream>>>(bufLin, src, dst, dinv, bufAgg, E);
  post_relu_bf16<<<(N * 32 + TB - 1) / TB, TB, 0, stream>>>(bufAgg, bufLin, dinv, b1, hb, N);

  // ---- layer 2 ----
  gemm_wmma<HID><<<gblocks, TB, 0, stream>>>(hb, W2T, bufLin, N);
  zero_f32<<<2048, TB, 0, stream>>>(bufAgg, N * HID);
  edge_agg<<<2048, TB, 0, stream>>>(bufLin, src, dst, dinv, bufAgg, E);
  post_pool<<<(N * 32 + TB - 1) / TB, TB, 0, stream>>>(bufAgg, bufLin, dinv, b2, batch, pool, cnt, N);

  // ---- head ----
  head<<<(GR * 32 + TB - 1) / TB, TB, 0, stream>>>(pool, cnt, Wlin, blin, out);
}
